// SwinTransformerBlock_64020782514967
// MI455X (gfx1250) — matrix operand order repaired and checked by reading, not compiled, not run on hardware
//
#include <hip/hip_runtime.h>

// ---------------------------------------------------------------------------
// Swin-V2 block for MI455X (gfx1250, wave32, WMMA).
// All GEMMs run on v_wmma_f32_16x16x32_f16 (f16 inputs, f32 accumulate).
// Dense GEMMs use 32x64 register tiles: 8 WMMA per 12 b128 loads per k-step.
// ---------------------------------------------------------------------------

typedef __attribute__((ext_vector_type(16))) _Float16 v16h;
typedef __attribute__((ext_vector_type(8)))  float    v8f;

#define B_     4
#define HW_    128
#define DIM_   256
#define HEADS_ 8
#define WS_    16
#define SS_    8
#define N_     256                 // tokens per window
#define NWIN_  64                  // windows per image
#define TOKENS 65536ull            // B * NW * N
#define HDIM_  1024

__device__ __forceinline__ v8f wmma32f16(v16h a, v16h b, v8f c) {
  // D = A(16x32,f16) x B(32x16,f16) + C(16x16,f32)
  return __builtin_amdgcn_wmma_f32_16x16x32_f16(false, a, false, b, (short)0, c,
                                                false, false);
}

union FragU { v16h v; uint4 q[2]; };

// A-matrix 16x32 f16 fragment (ISA 7.12.2):
// lane = 16*half + m ; VGPR v<4: K = half*8 + 2v..  ; v>=4: K = 16 + half*8 + ...
// => per lane two contiguous 16B loads at K = half*8 and K = 16 + half*8.
__device__ __forceinline__ v16h loadA(const _Float16* base, int ld, int row0, int k0) {
  int lane = (int)(threadIdx.x & 31);
  int m = lane & 15, half = lane >> 4;
  const _Float16* p = base + (size_t)(row0 + m) * ld + k0 + half * 8;
  FragU u;
  u.q[0] = *(const uint4*)(p);
  u.q[1] = *(const uint4*)(p + 16);
  return u.v;
}

// B-matrix 32x16 f16 fragment: lane = column n; same K order as loadA:
// two 16B loads at K = half*8 and K = 16 + half*8. Source is row-major
// W[n][k] (B = W^T).
__device__ __forceinline__ v16h loadB(const _Float16* base, int ld, int n0, int k0) {
  int lane = (int)(threadIdx.x & 31);
  int n = lane & 15, half = lane >> 4;
  const _Float16* p = base + (size_t)(n0 + n) * ld + k0 + half * 8;
  FragU u;
  u.q[0] = *(const uint4*)(p);
  u.q[1] = *(const uint4*)(p + 16);
  return u.v;
}

// windowed row r  <->  flat token s  (cyclic shift by SS + 16x16 partition);
// same mapping applies forward (partition) and reverse (merge).
__device__ __forceinline__ int unwindow_row(int r) {
  int b = r >> 14, w = (r >> 8) & 63, n = r & 255;
  int wy = w >> 3, wx = w & 7, ty = n >> 4, tx = n & 15;
  int y = (wy * 16 + ty + SS_) & 127;
  int x = (wx * 16 + tx + SS_) & 127;
  return (b << 14) + y * 128 + x;
}

// ---------------------------------------------------------------------------
__global__ void f32_to_f16_kernel(const float* __restrict__ s,
                                  _Float16* __restrict__ d, int n) {
  int t = blockIdx.x * 256 + threadIdx.x;
  if (t < n) d[t] = (_Float16)s[t];
}

// CPB: tbl[idx][h] = (relu(coords @ w1^T + b1) @ w2^T), idx in [0, 961)
__global__ void cpb_tbl_kernel(const float* __restrict__ w1,
                               const float* __restrict__ b1,
                               const float* __restrict__ w2,
                               float* __restrict__ tbl) {
  int t = blockIdx.x * 256 + threadIdx.x;
  if (t >= 961 * 8) return;
  int idx = t >> 3, h = t & 7;
  int i0 = idx / 31, i1 = idx % 31;
  float r0 = (float)(i0 - 15) * (8.0f / 15.0f);
  float r1 = (float)(i1 - 15) * (8.0f / 15.0f);
  float c0 = log2f(fabsf(r0) + 1.0f) * (1.0f / 3.0f); if (r0 < 0.f) c0 = -c0;
  float c1 = log2f(fabsf(r1) + 1.0f) * (1.0f / 3.0f); if (r1 < 0.f) c1 = -c1;
  float acc = 0.f;
  for (int j = 0; j < 512; ++j) {
    float hv = fmaxf(c0 * w1[2 * j] + c1 * w1[2 * j + 1] + b1[j], 0.f);
    acc += hv * w2[h * 512 + j];
  }
  tbl[idx * 8 + h] = acc;
}

// rpbsig[h][n][m] = 16 * sigmoid(tbl[relpos_idx(n,m)][h])
__global__ void rpb_kernel(const float* __restrict__ tbl,
                           float* __restrict__ rpbsig) {
  int t = blockIdx.x * 256 + threadIdx.x;   // 8 * 65536 total
  int h = t >> 16, nm = t & 65535, n = nm >> 8, m = nm & 255;
  int dr = (n >> 4) - (m >> 4) + 15;
  int dc = (n & 15) - (m & 15) + 15;
  float v = tbl[(dr * 31 + dc) * 8 + h];
  rpbsig[t] = 16.0f / (1.0f + __expf(-v));
}

// LayerNorm over 256 dims; one wave per row; optional shift+window permutation
// of the *source* row (permute=1 for LN1, 0 for LN2).
__global__ __launch_bounds__(256) void ln_kernel(const float* __restrict__ src,
                                                 _Float16* __restrict__ dst,
                                                 const float* __restrict__ g,
                                                 const float* __restrict__ bt,
                                                 int permute) {
  int wave = threadIdx.x >> 5, lane = threadIdx.x & 31;
  int r = blockIdx.x * 8 + wave;
  int s = permute ? unwindow_row(r) : r;
  const float* xr = src + (size_t)s * DIM_;
  float v[8], sum = 0.f;
#pragma unroll
  for (int i = 0; i < 8; ++i) { v[i] = xr[lane + 32 * i]; sum += v[i]; }
  sum += __shfl_xor(sum, 1, 32);  sum += __shfl_xor(sum, 2, 32);
  sum += __shfl_xor(sum, 4, 32);  sum += __shfl_xor(sum, 8, 32);
  sum += __shfl_xor(sum, 16, 32);
  float mu = sum * (1.0f / 256.0f), var = 0.f;
#pragma unroll
  for (int i = 0; i < 8; ++i) { float d = v[i] - mu; var += d * d; }
  var += __shfl_xor(var, 1, 32);  var += __shfl_xor(var, 2, 32);
  var += __shfl_xor(var, 4, 32);  var += __shfl_xor(var, 8, 32);
  var += __shfl_xor(var, 16, 32);
  float rs = rsqrtf(var * (1.0f / 256.0f) + 1e-5f);
#pragma unroll
  for (int i = 0; i < 8; ++i) {
    int c = lane + 32 * i;
    dst[(size_t)r * DIM_ + c] = (_Float16)((v[i] - mu) * rs * g[c] + bt[c]);
  }
}

// QKV GEMM (65536x256 @ 256x768^T) + bias + L2-norm(q,k) + scatter.
// One wave computes a 32x64 tile: 8 WMMA per 12 b128 loads per k-step.
__global__ __launch_bounds__(32) void qkv_gemm_kernel(
    const _Float16* __restrict__ Xw, const _Float16* __restrict__ Wq,
    const float* __restrict__ qbias, const float* __restrict__ vbias,
    _Float16* __restrict__ Qm, _Float16* __restrict__ Km,
    _Float16* __restrict__ Vt) {
  int row0 = blockIdx.x * 32;
  int c0   = blockIdx.y * 64;            // 12 col groups over 768
  int lane = threadIdx.x & 31, nn = lane & 15, half = lane >> 4;
  v8f acc[2][4] = {};
  for (int k0 = 0; k0 < DIM_; k0 += 32) {
    v16h a0 = loadA(Xw, DIM_, row0, k0);
    v16h a1 = loadA(Xw, DIM_, row0 + 16, k0);
    v16h b0 = loadB(Wq, DIM_, c0, k0);
    v16h b1 = loadB(Wq, DIM_, c0 + 16, k0);
    v16h b2 = loadB(Wq, DIM_, c0 + 32, k0);
    v16h b3 = loadB(Wq, DIM_, c0 + 48, k0);
    acc[0][0] = wmma32f16(a0, b0, acc[0][0]);
    acc[0][1] = wmma32f16(a0, b1, acc[0][1]);
    acc[0][2] = wmma32f16(a0, b2, acc[0][2]);
    acc[0][3] = wmma32f16(a0, b3, acc[0][3]);
    acc[1][0] = wmma32f16(a1, b0, acc[1][0]);
    acc[1][1] = wmma32f16(a1, b1, acc[1][1]);
    acc[1][2] = wmma32f16(a1, b2, acc[1][2]);
    acc[1][3] = wmma32f16(a1, b3, acc[1][3]);
  }
  int which = c0 >> 8;                   // 0=q, 1=k, 2=v (c0%256 aligned)
  int headbase = (c0 & 255) >> 5;
#pragma unroll
  for (int cp = 0; cp < 2; ++cp) {       // two 32-col head slices
    int head = headbase + cp;
    int cc = (c0 & 255) + cp * 32;
    float bias0 = 0.f, bias1 = 0.f;
    if (which == 0) { bias0 = qbias[cc + nn]; bias1 = qbias[cc + 16 + nn]; }
    if (which == 2) { bias0 = vbias[cc + nn]; bias1 = vbias[cc + 16 + nn]; }
#pragma unroll
    for (int r = 0; r < 2; ++r) {
      v8f A0 = acc[r][2 * cp], A1 = acc[r][2 * cp + 1];
      float inv[8];
#pragma unroll
      for (int j = 0; j < 8; ++j) {
        float x0 = A0[j] + bias0, x1 = A1[j] + bias1;
        A0[j] = x0; A1[j] = x1;
        if (which < 2) {                 // cosine attention: L2 norm over d=32
          float ss = x0 * x0 + x1 * x1;
          ss += __shfl_xor(ss, 1, 32); ss += __shfl_xor(ss, 2, 32);
          ss += __shfl_xor(ss, 4, 32); ss += __shfl_xor(ss, 8, 32);
          inv[j] = 1.0f / fmaxf(sqrtf(ss), 1e-12f);
        } else inv[j] = 1.0f;
      }
#pragma unroll
      for (int j = 0; j < 8; ++j) {
        int tok = row0 + r * 16 + j + 8 * half;
        int win = tok >> 8, n = tok & 255;
        size_t wh = (size_t)(win * HEADS_ + head);
        if (which == 2) {                // V stored transposed: Vt[wh][d][key]
          Vt[(wh * 32 + nn) * 256 + n]      = (_Float16)A0[j];
          Vt[(wh * 32 + 16 + nn) * 256 + n] = (_Float16)A1[j];
        } else {
          _Float16* dst = (which == 0) ? Qm : Km;   // [wh][token][d]
          size_t base = (wh * 256 + (size_t)n) * 32;
          dst[base + nn]      = (_Float16)(A0[j] * inv[j]);
          dst[base + 16 + nn] = (_Float16)(A1[j] * inv[j]);
        }
      }
    }
  }
}

// Flash-style attention per (window, head). 8 waves, each owns 2 q-tiles of 16
// rows; online softmax; P goes through per-wave LDS to become an A-fragment.
__global__ __launch_bounds__(256) void attn_kernel(
    const _Float16* __restrict__ Qm, const _Float16* __restrict__ Km,
    const _Float16* __restrict__ Vt, const float* __restrict__ rpbsig,
    const float* __restrict__ logit_scale, _Float16* __restrict__ attn_out) {
  __shared__ _Float16 Pbuf[8][16][32];
  int wh = blockIdx.x;                       // win*8 + head
  int win = wh >> 3, head = wh & 7;
  int w64 = win & 63, wy = w64 >> 3, wx = w64 & 7;
  int wave = threadIdx.x >> 5, lane = threadIdx.x & 31;
  int nn = lane & 15, half = lane >> 4;
  float scale = __expf(fminf(logit_scale[head], 4.6051702f));  // ln(100)
  const _Float16* qb = Qm + (size_t)wh * 256 * 32;
  const _Float16* kb = Km + (size_t)wh * 256 * 32;
  const _Float16* vb = Vt + (size_t)wh * 32 * 256;
  const float* bias = rpbsig + (size_t)head * 65536;
  for (int qt = wave; qt < 16; qt += 8) {
    v16h aQ = loadA(qb, 32, qt * 16, 0);     // q-tile fragment, reused all steps
    v8f o0 = {}, o1 = {};
    float mrow[8], lrow[8];
#pragma unroll
    for (int j = 0; j < 8; ++j) { mrow[j] = -3.0e38f; lrow[j] = 0.f; }
    for (int kt = 0; kt < 256; kt += 32) {
      v16h bk0 = loadB(kb, 32, kt, 0);
      v16h bk1 = loadB(kb, 32, kt + 16, 0);
      v8f z = {};
      v8f s0 = wmma32f16(aQ, bk0, z);
      v8f s1 = wmma32f16(aQ, bk1, z);
      int key0 = kt + nn, key1 = kt + 16 + nn;
      int k0ry = (wy < 7) ? 0 : (((key0 >> 4) < 8) ? 1 : 2);
      int k0rx = (wx < 7) ? 0 : (((key0 & 15) < 8) ? 1 : 2);
      int k1ry = (wy < 7) ? 0 : (((key1 >> 4) < 8) ? 1 : 2);
      int k1rx = (wx < 7) ? 0 : (((key1 & 15) < 8) ? 1 : 2);
#pragma unroll
      for (int j = 0; j < 8; ++j) {
        int qrow = qt * 16 + j + 8 * half;
        int qry = (wy < 7) ? 0 : (((qrow >> 4) < 8) ? 1 : 2);
        int qrx = (wx < 7) ? 0 : (((qrow & 15) < 8) ? 1 : 2);
        float e0 = s0[j] * scale + bias[qrow * 256 + key0] +
                   ((qry == k0ry && qrx == k0rx) ? 0.f : -100.f);
        float e1 = s1[j] * scale + bias[qrow * 256 + key1] +
                   ((qry == k1ry && qrx == k1rx) ? 0.f : -100.f);
        float t = fmaxf(e0, e1);
        t = fmaxf(t, __shfl_xor(t, 1, 32));
        t = fmaxf(t, __shfl_xor(t, 2, 32));
        t = fmaxf(t, __shfl_xor(t, 4, 32));
        t = fmaxf(t, __shfl_xor(t, 8, 32));
        float mnew = fmaxf(mrow[j], t);
        float alpha = __expf(mrow[j] - mnew);
        float p0 = __expf(e0 - mnew), p1 = __expf(e1 - mnew);
        float ps = p0 + p1;
        ps += __shfl_xor(ps, 1, 32); ps += __shfl_xor(ps, 2, 32);
        ps += __shfl_xor(ps, 4, 32); ps += __shfl_xor(ps, 8, 32);
        lrow[j] = lrow[j] * alpha + ps;
        mrow[j] = mnew;
        o0[j] *= alpha; o1[j] *= alpha;
        Pbuf[wave][j + 8 * half][nn]      = (_Float16)p0;
        Pbuf[wave][j + 8 * half][16 + nn] = (_Float16)p1;
      }
      v16h aP  = loadA(&Pbuf[wave][0][0], 32, 0, 0);   // same-wave LDS bounce
      v16h bv0 = loadB(vb, 256, 0, kt);                // V^T rows = d
      v16h bv1 = loadB(vb, 256, 16, kt);
      o0 = wmma32f16(aP, bv0, o0);
      o1 = wmma32f16(aP, bv1, o1);
    }
#pragma unroll
    for (int j = 0; j < 8; ++j) {
      float iv = 1.0f / lrow[j];
      int tok = win * 256 + qt * 16 + j + 8 * half;
      size_t base = (size_t)tok * DIM_ + head * 32;
      attn_out[base + nn]      = (_Float16)(o0[j] * iv);
      attn_out[base + 16 + nn] = (_Float16)(o1[j] * iv);
    }
  }
}

// proj GEMM (32x64 tile) + bias + window-reverse/un-shift + residual -> d_out
__global__ __launch_bounds__(32) void proj_kernel(
    const _Float16* __restrict__ att, const _Float16* __restrict__ Wp,
    const float* __restrict__ pb, const float* __restrict__ x_in,
    float* __restrict__ out) {
  int row0 = blockIdx.x * 32, c0 = blockIdx.y * 64;
  int lane = threadIdx.x & 31, nn = lane & 15, half = lane >> 4;
  v8f acc[2][4] = {};
  for (int k0 = 0; k0 < DIM_; k0 += 32) {
    v16h a0 = loadA(att, DIM_, row0, k0);
    v16h a1 = loadA(att, DIM_, row0 + 16, k0);
    v16h b0 = loadB(Wp, DIM_, c0, k0);
    v16h b1 = loadB(Wp, DIM_, c0 + 16, k0);
    v16h b2 = loadB(Wp, DIM_, c0 + 32, k0);
    v16h b3 = loadB(Wp, DIM_, c0 + 48, k0);
    acc[0][0] = wmma32f16(a0, b0, acc[0][0]);
    acc[0][1] = wmma32f16(a0, b1, acc[0][1]);
    acc[0][2] = wmma32f16(a0, b2, acc[0][2]);
    acc[0][3] = wmma32f16(a0, b3, acc[0][3]);
    acc[1][0] = wmma32f16(a1, b0, acc[1][0]);
    acc[1][1] = wmma32f16(a1, b1, acc[1][1]);
    acc[1][2] = wmma32f16(a1, b2, acc[1][2]);
    acc[1][3] = wmma32f16(a1, b3, acc[1][3]);
  }
#pragma unroll
  for (int cp = 0; cp < 4; ++cp) {
    int cc = c0 + cp * 16;
    float bb = pb[cc + nn];
#pragma unroll
    for (int r = 0; r < 2; ++r) {
#pragma unroll
      for (int j = 0; j < 8; ++j) {
        int rw = row0 + r * 16 + j + 8 * half;
        int s = unwindow_row(rw);
        size_t o = (size_t)s * DIM_ + cc + nn;
        out[o] = x_in[o] + acc[r][cp][j] + bb;
      }
    }
  }
}

// Fused MLP: fc1 (16x64 tiles) -> GELU -> LDS f16 tile -> fc2 -> residual add.
__global__ __launch_bounds__(256) void mlp_kernel(
    const _Float16* __restrict__ H2, const _Float16* __restrict__ W1,
    const float* __restrict__ b1, const _Float16* __restrict__ W2,
    const float* __restrict__ b2, float* __restrict__ out) {
  __shared__ _Float16 mid[16][HDIM_];        // 32 KB
  int row0 = blockIdx.x * 16;
  int wave = threadIdx.x >> 5, lane = threadIdx.x & 31;
  int nn = lane & 15, half = lane >> 4;
  for (int cg = wave; cg < 16; cg += 8) {    // 1024 fc1 outputs, 64 per group
    int c0 = cg * 64;
    v8f acc[4] = {};
    for (int k0 = 0; k0 < DIM_; k0 += 32) {
      v16h a  = loadA(H2, DIM_, row0, k0);
      v16h w0 = loadB(W1, DIM_, c0, k0);
      v16h w1 = loadB(W1, DIM_, c0 + 16, k0);
      v16h w2 = loadB(W1, DIM_, c0 + 32, k0);
      v16h w3 = loadB(W1, DIM_, c0 + 48, k0);
      acc[0] = wmma32f16(a, w0, acc[0]);
      acc[1] = wmma32f16(a, w1, acc[1]);
      acc[2] = wmma32f16(a, w2, acc[2]);
      acc[3] = wmma32f16(a, w3, acc[3]);
    }
#pragma unroll
    for (int cp = 0; cp < 4; ++cp) {
      int cc = c0 + cp * 16;
      float bb = b1[cc + nn];
#pragma unroll
      for (int j = 0; j < 8; ++j) {
        float v0 = acc[cp][j] + bb;
        v0 = 0.5f * v0 * (1.0f + erff(v0 * 0.70710678f));   // exact GELU
        mid[j + 8 * half][cc + nn] = (_Float16)v0;
      }
    }
  }
  __syncthreads();
  int c0 = wave * 32;                        // 256 fc2 outputs, 32 per wave
  v8f a0 = {}, a1 = {};
  for (int k0 = 0; k0 < HDIM_; k0 += 32) {
    v16h a  = loadA(&mid[0][0], HDIM_, 0, k0);
    v16h w0 = loadB(W2, HDIM_, c0, k0);
    v16h w1 = loadB(W2, HDIM_, c0 + 16, k0);
    a0 = wmma32f16(a, w0, a0);
    a1 = wmma32f16(a, w1, a1);
  }
  float bb0 = b2[c0 + nn], bb1 = b2[c0 + 16 + nn];
#pragma unroll
  for (int j = 0; j < 8; ++j) {
    int r = row0 + j + 8 * half;
    size_t o = (size_t)r * DIM_ + c0;
    out[o + nn]      += a0[j] + bb0;
    out[o + 16 + nn] += a1[j] + bb1;
  }
}

// ---------------------------------------------------------------------------
extern "C" void kernel_launch(void* const* d_in, const int* in_sizes, int n_in,
                              void* d_out, int out_size, void* d_ws, size_t ws_size,
                              hipStream_t stream) {
  (void)in_sizes; (void)n_in; (void)out_size; (void)ws_size;
  const float* x        = (const float*)d_in[0];
  const float* qkv_w    = (const float*)d_in[1];
  const float* q_bias   = (const float*)d_in[2];
  const float* v_bias   = (const float*)d_in[3];
  const float* logit_sc = (const float*)d_in[4];
  const float* cpb_w1   = (const float*)d_in[5];
  const float* cpb_b1   = (const float*)d_in[6];
  const float* cpb_w2   = (const float*)d_in[7];
  const float* proj_w   = (const float*)d_in[8];
  const float* proj_b   = (const float*)d_in[9];
  const float* n1g      = (const float*)d_in[10];
  const float* n1b      = (const float*)d_in[11];
  const float* n2g      = (const float*)d_in[12];
  const float* n2b      = (const float*)d_in[13];
  const float* fc1_w    = (const float*)d_in[14];
  const float* fc1_b    = (const float*)d_in[15];
  const float* fc2_w    = (const float*)d_in[16];
  const float* fc2_b    = (const float*)d_in[17];
  float* out = (float*)d_out;

  // Workspace carve (~138 MB). Region0 is reused Xw -> attn_out -> H2.
  char* p = (char*)d_ws;
  auto take = [&](size_t bytes) {
    char* r = p; p += (bytes + 255) & ~(size_t)255; return r;
  };
  _Float16* Xw      = (_Float16*)take(TOKENS * DIM_ * 2);   // reused 3x
  _Float16* Qm      = (_Float16*)take(TOKENS * DIM_ * 2);
  _Float16* Km      = (_Float16*)take(TOKENS * DIM_ * 2);
  _Float16* Vt      = (_Float16*)take(TOKENS * DIM_ * 2);
  _Float16* qkvw16  = (_Float16*)take(768 * 256 * 2);
  _Float16* projw16 = (_Float16*)take(256 * 256 * 2);
  _Float16* fc1w16  = (_Float16*)take(1024 * 256 * 2);
  _Float16* fc2w16  = (_Float16*)take(256 * 1024 * 2);
  float*    tbl     = (float*)take(961 * 8 * 4);
  float*    rpbsig  = (float*)take(8 * 65536 * 4);

  // Weight conversion + CPB tables (independent, cheap)
  f32_to_f16_kernel<<<768, 256, 0, stream>>>(qkv_w, qkvw16, 768 * 256);
  f32_to_f16_kernel<<<256, 256, 0, stream>>>(proj_w, projw16, 256 * 256);
  f32_to_f16_kernel<<<1024, 256, 0, stream>>>(fc1_w, fc1w16, 1024 * 256);
  f32_to_f16_kernel<<<1024, 256, 0, stream>>>(fc2_w, fc2w16, 256 * 1024);
  cpb_tbl_kernel<<<31, 256, 0, stream>>>(cpb_w1, cpb_b1, cpb_w2, tbl);
  rpb_kernel<<<2048, 256, 0, stream>>>(tbl, rpbsig);

  // LN1 + cyclic shift + window partition -> f16
  ln_kernel<<<8192, 256, 0, stream>>>(x, Xw, n1g, n1b, 1);
  // QKV GEMM + bias + L2-norm + layout scatter (32x64 tiles)
  qkv_gemm_kernel<<<dim3(2048, 12), 32, 0, stream>>>(Xw, qkvw16, q_bias, v_bias,
                                                     Qm, Km, Vt);
  // attention (reuses Xw region as attn_out)
  attn_kernel<<<2048, 256, 0, stream>>>(Qm, Km, Vt, rpbsig, logit_sc, Xw);
  // proj + window reverse + un-shift + residual -> d_out (32x64 tiles)
  proj_kernel<<<dim3(2048, 4), 32, 0, stream>>>(Xw, projw16, proj_b, x, out);
  // LN2 (reuses Xw region as H2)
  ln_kernel<<<8192, 256, 0, stream>>>(out, Xw, n2g, n2b, 0);
  // fused MLP, accumulates into d_out
  mlp_kernel<<<4096, 256, 0, stream>>>(Xw, fc1w16, fc1_b, fc2w16, fc2_b, out);
}
